// GCNConvDiffPool_54606214201548
// MI455X (gfx1250) — matrix operand order repaired and checked by reading, not compiled, not run on hardware
//
#include <hip/hip_runtime.h>

#define N_NODES 16384
#define E_EDGES 1048576
#define C_IN    128

typedef __attribute__((ext_vector_type(16))) _Float16 v16h;
typedef __attribute__((ext_vector_type(8)))  float    v8f;

// ---- graph-norm setup --------------------------------------------------
__global__ __launch_bounds__(256) void k_init(float* __restrict__ deg,
                                              float* __restrict__ acc) {
  int i = blockIdx.x * 256 + threadIdx.x;
  deg[i] = 1.0f;   // self-loop weight
  acc[i] = 0.0f;
}

__global__ __launch_bounds__(256) void k_deg(const int* __restrict__ col,
                                             const float* __restrict__ ew,
                                             float* __restrict__ deg) {
  int e = blockIdx.x * 256 + threadIdx.x;
  atomicAdd(&deg[col[e]], ew[e]);
}

__global__ __launch_bounds__(256) void k_dinv(float* __restrict__ deg) {
  int i = blockIdx.x * 256 + threadIdx.x;
  float d = deg[i];
  deg[i] = d > 0.0f ? rsqrtf(d) : 0.0f;   // deg buffer becomes dinv
}

// ---- layer-1 GEMV via WMMA: h[i] = dot(x[i,:], W1) ---------------------
// One wave per 16-row tile; K=128 consumed as 4x v_wmma_f32_16x16x32_f16.
// B replicates W1 in all 16 columns, so every D column equals h.
__global__ __launch_bounds__(256) void k_gemv_wmma(const float* __restrict__ x,
                                                   const float* __restrict__ W1,
                                                   float* __restrict__ h) {
  __shared__ _Float16 wsh[C_IN];
  int t = threadIdx.x;
  if (t < C_IN) wsh[t] = (_Float16)W1[t];
  __syncthreads();

  int wave = t >> 5;
  int lane = t & 31;
  int half = lane >> 4;                 // lane half selects K sub-ranges
  int tile = blockIdx.x * 8 + wave;     // 16 rows per wave
  int row  = tile * 16 + (lane & 15);   // A layout: lane = row M

  const float* xr = x + (size_t)row * C_IN;
  v8f c = {};
  #pragma unroll
  for (int kb = 0; kb < C_IN; kb += 32) {
    v16h a, b;
    // 16-bit A 16x32 layout: VGPR0-3 hold K = half*8 + 0..7,
    //                        VGPR4-7 hold K = 16 + half*8 + 0..7
    #pragma unroll
    for (int i = 0; i < 8; ++i) {
      a[i]     = (_Float16)xr[kb + half * 8 + i];
      a[8 + i] = (_Float16)xr[kb + 16 + half * 8 + i];
    }
    // 16-bit B 32x16: lane L holds column N=L&15, K = (L>>4)*8 + 0..7, 16 + (L>>4)*8 + 0..7
    // (replicated across columns -> broadcast reads from LDS)
    #pragma unroll
    for (int i = 0; i < 16; ++i)
      b[i] = wsh[kb + (i < 8 ? half * 8 + i : 16 + half * 8 + (i - 8))];

    c = __builtin_amdgcn_wmma_f32_16x16x32_f16(
        /*neg_a=*/false, a, /*neg_b=*/false, b,
        /*c_mod=*/(short)0, c, /*reuse_a=*/false, /*reuse_b=*/false);
  }
  // f32 C/D layout: VGPR g, lane L -> M = 8*(L>>4)+g, N = L&15.
  // All columns identical; column-0 lanes write the 16 rows.
  if ((lane & 15) == 0) {
    #pragma unroll
    for (int g = 0; g < 8; ++g)
      h[tile * 16 + half * 8 + g] = c[g];
  }
}

// ---- edge scatter: acc[col] += h[row] * dinv[row]*ew*dinv[col] ---------
__global__ __launch_bounds__(256) void k_scatter(const int* __restrict__ row,
                                                 const int* __restrict__ col,
                                                 const float* __restrict__ ew,
                                                 const float* __restrict__ dinv,
                                                 const float* __restrict__ h,
                                                 float* __restrict__ acc) {
  int e = blockIdx.x * 256 + threadIdx.x;
  int r = row[e], ci = col[e];
  float v = dinv[r] * ew[e] * dinv[ci] * h[r];
  atomicAdd(&acc[ci], v);
}

// ---- layer-1 self-loop + bias, layer-2 linear; reset acc ---------------
__global__ __launch_bounds__(256) void k_mid(const float* __restrict__ dinv,
                                             float* __restrict__ h,
                                             float* __restrict__ acc,
                                             const float* __restrict__ b1,
                                             const float* __restrict__ W2) {
  int i = blockIdx.x * 256 + threadIdx.x;
  float di = dinv[i];
  float o1 = acc[i] + h[i] * di * di + b1[0];   // self-loop norm = dinv^2
  h[i]   = o1 * W2[0];                          // layer-2 linear (1->1)
  acc[i] = 0.0f;                                // re-zero accumulator
}

// ---- layer-2 self-loop + bias, emit output + reg -----------------------
__global__ __launch_bounds__(256) void k_final(const float* __restrict__ dinv,
                                               const float* __restrict__ h,
                                               const float* __restrict__ acc,
                                               const float* __restrict__ b2,
                                               float* __restrict__ out) {
  int i = blockIdx.x * 256 + threadIdx.x;
  float di = dinv[i];
  out[i] = acc[i] + h[i] * di * di + b2[0];
  if (i == 0) out[N_NODES] = 0.0f;              // reg = zeros((1,))
}

extern "C" void kernel_launch(void* const* d_in, const int* in_sizes, int n_in,
                              void* d_out, int out_size, void* d_ws, size_t ws_size,
                              hipStream_t stream) {
  const float* x  = (const float*)d_in[0];   // [N,128]
  const int*   ei = (const int*)  d_in[1];   // [2,E]
  const float* ea = (const float*)d_in[2];   // [E,1] -> [E]
  /* d_in[3] = adj, unused */
  const float* W1 = (const float*)d_in[4];   // [128,1]
  const float* b1 = (const float*)d_in[5];   // [1]
  const float* W2 = (const float*)d_in[6];   // [1,1]
  const float* b2 = (const float*)d_in[7];   // [1]
  float* out = (float*)d_out;                // N+1 floats

  float* ws   = (float*)d_ws;
  float* dinv = ws;                 // N: degree, then rsqrt in place
  float* h    = ws + N_NODES;       // N: per-layer node feature
  float* acc  = ws + 2 * N_NODES;   // N: segment-sum accumulator

  const int* row = ei;              // source nodes j
  const int* col = ei + E_EDGES;    // target nodes i

  dim3 blk(256);
  k_init <<<N_NODES / 256, blk, 0, stream>>>(dinv, acc);
  k_deg  <<<E_EDGES / 256, blk, 0, stream>>>(col, ea, dinv);
  k_dinv <<<N_NODES / 256, blk, 0, stream>>>(dinv);
  k_gemv_wmma<<<N_NODES / 128, blk, 0, stream>>>(x, W1, h);
  k_scatter<<<E_EDGES / 256, blk, 0, stream>>>(row, col, ea, dinv, h, acc);
  k_mid  <<<N_NODES / 256, blk, 0, stream>>>(dinv, h, acc, b1, W2);
  k_scatter<<<E_EDGES / 256, blk, 0, stream>>>(row, col, ea, dinv, h, acc);
  k_final<<<N_NODES / 256, blk, 0, stream>>>(dinv, h, acc, b2, out);
}
